// ALSHConv2d_56014963474550
// MI455X (gfx1250) — matrix operand order repaired and checked by reading, not compiled, not run on hardware
//
#include <hip/hip_runtime.h>

// ---------------------------------------------------------------------------
// ALSH-gated 3x3 conv, implicit GEMM on CDNA5 WMMA (bf16 in, f32 accum).
//   M = O = 512, N = 32*56*56 = 100352, K = 9*256 = 2304
//   Block tile 256(M) x 128(N), K-step 32, 256 threads = 8 wave32 (4x2 grid),
//   each wave: 4x4 tiles -> 16 x v_wmma_f32_16x16x32_bf16 per K-step.
//
// v3: weights pre-converted once into a bf16 *fragment-ready* record layout
//     in d_ws (record = one lane's 16 bf16 = 32B, indexed [mtile][step][lane]).
//     A fragments are then loaded straight from global (L2-resident, coalesced
//     b128); only the im2col B tile goes through LDS (double-buffered 2x8KB,
//     one barrier per K-step). Buckets in d_ws; gate fused into epilogue.
// Workspace needed: 4KB + 32*72*32*32B = ~2.37 MB.
// ---------------------------------------------------------------------------

typedef __attribute__((ext_vector_type(16))) __bf16 v16bf;
typedef __attribute__((ext_vector_type(2)))  __bf16 v2bf;
typedef __attribute__((ext_vector_type(8)))  float  v8f;

#define CN   32
#define CC   256
#define CWID 56
#define CHW  3136      // 56*56
#define CO   512
#define CD   2304      // 9*256
#define NSTEP 72       // 9 positions * (256/32) channel chunks
#define NMT  32        // 512/16 global M tiles

__device__ __forceinline__ unsigned pack_bf16(float lo, float hi) {
#if defined(__has_builtin) && __has_builtin(__builtin_amdgcn_cvt_pk_bf16_f32)
  v2bf r = __builtin_amdgcn_cvt_pk_bf16_f32(lo, hi);
  return __builtin_bit_cast(unsigned, r);
#else
  unsigned ul = __float_as_uint(lo);
  unsigned uh = __float_as_uint(hi);
  ul = (ul + 0x7fffu + ((ul >> 16) & 1u)) >> 16;           // RNE
  uh = ((uh + 0x7fffu + ((uh >> 16) & 1u)) >> 16) << 16;
  return ul | uh;
#endif
}

__device__ __forceinline__ int bucket16(float d) {
  return ((int)fabsf(floorf(d))) & 15;   // mod(abs(floor(d)), 16)
}

// ---------------- one-time A prep: f32 weights -> fragment-ready bf16 ------
// Record r = (mt*NSTEP + s)*32 + lane; lane<16 holds row o=mt*16+lane,
// K{0..15} of chunk 32*s; lane>=16 holds same row, K{16..31}.

__global__ void aprep_kernel(const float* __restrict__ kern,
                             uint4* __restrict__ wsA4) {
  const int r    = blockIdx.x * 256 + threadIdx.x;   // 0..73727
  const int lane = r & 31;
  const int ms   = r >> 5;            // mt*NSTEP + s
  const int s    = ms % NSTEP;
  const int mt   = ms / NSTEP;
  const int o    = mt * 16 + (lane & 15);
  const int hf   = lane >> 4;
  const float* kp = kern + (size_t)o * CD + 32 * s + hf * 16;
  float a[16];
  #pragma unroll
  for (int j = 0; j < 8; ++j) { a[j] = kp[j]; a[8 + j] = kp[8 + j]; }
  uint4 u0, u1;
  u0.x = pack_bf16(a[0],  a[1]);  u0.y = pack_bf16(a[2],  a[3]);
  u0.z = pack_bf16(a[4],  a[5]);  u0.w = pack_bf16(a[6],  a[7]);
  u1.x = pack_bf16(a[8],  a[9]);  u1.y = pack_bf16(a[10], a[11]);
  u1.z = pack_bf16(a[12], a[13]); u1.w = pack_bf16(a[14], a[15]);
  wsA4[(size_t)r * 2]     = u0;
  wsA4[(size_t)r * 2 + 1] = u1;
}

// ---------------- bucket kernels (negligible cost) -------------------------

__global__ void kbucket_kernel(const float* __restrict__ kern,
                               const float* __restrict__ hash_a,
                               int* __restrict__ kb) {
  const int o = blockIdx.x, t = threadIdx.x;
  const float* kp = kern + (size_t)o * CD;
  float s2 = 0.f, sd = 0.f;
  for (int i = t; i < CD; i += 256) {
    float v = kp[i];
    s2 += v * v;
    sd += v * hash_a[i];
  }
  __shared__ float r1[256], r2[256];
  r1[t] = s2; r2[t] = sd;
  __syncthreads();
  for (int off = 128; off > 0; off >>= 1) {
    if (t < off) { r1[t] += r1[t + off]; r2[t] += r2[t + off]; }
    __syncthreads();
  }
  if (t == 0) {
    float n2 = r1[0], dot = r2[0];
    float a0 = hash_a[CD], a1 = hash_a[CD + 1], a2 = hash_a[CD + 2];
    float n4 = n2 * n2, n8 = n4 * n4;           // n2^(2^i), i=0..2
    kb[o] = bucket16(dot + n2 * a0 + n4 * a1 + n8 * a2);
  }
}

__global__ void qbucket_kernel(const float* __restrict__ x,
                               const float* __restrict__ hash_a,
                               int* __restrict__ qb) {
  const int n = blockIdx.x, t = threadIdx.x;   // t = channel
  const float4* xp = reinterpret_cast<const float4*>(x + ((size_t)n * CC + t) * CHW);
  float s = 0.f;
  for (int i = 0; i < CHW / 4; ++i) {
    float4 v = xp[i];
    s += v.x + v.y + v.z + v.w;
  }
  float mean = s * (1.0f / (float)CHW);
  float asum = 0.f;
  #pragma unroll
  for (int j = 0; j < 9; ++j) asum += hash_a[j * CC + t];
  __shared__ float red[256];
  red[t] = mean * asum;
  __syncthreads();
  for (int off = 128; off > 0; off >>= 1) {
    if (t < off) red[t] += red[t + off];
    __syncthreads();
  }
  if (t == 0) {
    float aug = hash_a[CD] + hash_a[CD + 1] + hash_a[CD + 2];
    qb[n] = bucket16(red[0] + 0.5f * aug);
  }
}

// ---------------- main implicit-GEMM conv ----------------------------------

union FragU { uint4 u[2]; v16bf h; };

__launch_bounds__(256)
__global__ void alsh_conv_wmma(const float* __restrict__ x,
                               const uint4* __restrict__ wsA4,
                               const int*   __restrict__ kb,
                               const int*   __restrict__ qb,
                               float* __restrict__ out) {
  __shared__ uint4 ldsB[1024];   // double-buffered: 2 x (8 N-tiles * 32 * 2)

  const int t      = threadIdx.x;
  const int lane   = t & 31;
  const int wid    = t >> 5;
  const int wave_m = wid & 3;   // 0..3 -> M tiles 4*wave_m..+3
  const int wave_n = wid >> 2;  // 0..1 -> N tiles 4*wave_n..+3

  const int m0 = blockIdx.y * 256;   // output-channel base
  const int p0 = blockIdx.x * 128;   // flattened pixel base (n*3136+h*56+w)

  // ---- A fragment pointers (global, fragment-ready records) ----
  const int mtG0 = blockIdx.y * 16 + wave_m * 4;
  const uint4* aPtr[4];
  #pragma unroll
  for (int i = 0; i < 4; ++i)
    aPtr[i] = wsA4 + ((size_t)(mtG0 + i) * NSTEP * 32 + lane) * 2;

  // ---- B loader: thread -> (pixel, channel-group of 16) ----
  const int pl = t & 127;
  const int cg = t >> 7;         // 0..1
  const int p    = p0 + pl;
  const int nimg = p / CHW;
  const int phw  = p - nimg * CHW;
  const int h    = phw / CWID;
  const int w    = phw - h * CWID;
  const int bRec = ((pl >> 4) * 32 + (pl & 15) + 16 * cg) * 2;
  const float* xb = x + ((size_t)nimg * CC + cg * 16) * CHW + phw;

  unsigned vmask = 0;                           // 9-bit border validity
  #pragma unroll
  for (int pos = 0; pos < 9; ++pos) {
    const int ih = h + pos / 3 - 1, iw = w + pos % 3 - 1;
    if ((unsigned)ih < 56u && (unsigned)iw < 56u) vmask |= (1u << pos);
  }

  float breg[16];
  bool  bvalid = false;

  auto load_stepB = [&](int s) {
    const int pos = s >> 3;              // 0..8  = kh*3+kw
    const int c0  = (s & 7) << 5;        // channel chunk
    bvalid = (vmask >> pos) & 1u;
    const int posoff = (pos / 3) * CWID + (pos - (pos / 3) * 3) - (CWID + 1);
    const float* bp = xb + (size_t)c0 * CHW + (bvalid ? posoff : 0);
    #pragma unroll
    for (int j = 0; j < 16; ++j)
      breg[j] = bp[(size_t)j * CHW];
  };

  auto storeB = [&](int buf) {
    uint4* dst = ldsB + buf * 512 + bRec;
    uint4 u;
    u.x = bvalid ? pack_bf16(breg[0],  breg[1])  : 0u;
    u.y = bvalid ? pack_bf16(breg[2],  breg[3])  : 0u;
    u.z = bvalid ? pack_bf16(breg[4],  breg[5])  : 0u;
    u.w = bvalid ? pack_bf16(breg[6],  breg[7])  : 0u;
    dst[0] = u;
    u.x = bvalid ? pack_bf16(breg[8],  breg[9])  : 0u;
    u.y = bvalid ? pack_bf16(breg[10], breg[11]) : 0u;
    u.z = bvalid ? pack_bf16(breg[12], breg[13]) : 0u;
    u.w = bvalid ? pack_bf16(breg[14], breg[15]) : 0u;
    dst[1] = u;
  };

  auto prefetchA = [&](int s, FragU (&dst)[4]) {
    #pragma unroll
    for (int i = 0; i < 4; ++i) {
      const uint4* pp = aPtr[i] + (size_t)s * 64;   // 32 records * 2 uint4
      dst[i].u[0] = pp[0]; dst[i].u[1] = pp[1];
    }
  };

  v8f acc[4][4];
  #pragma unroll
  for (int i = 0; i < 4; ++i)
    #pragma unroll
    for (int j = 0; j < 4; ++j)
      acc[i][j] = (v8f){0.f, 0.f, 0.f, 0.f, 0.f, 0.f, 0.f, 0.f};

  FragU aF0[4], aF1[4];

  auto body = [&](int s, FragU (&aCur)[4], FragU (&aNext)[4], int bufCur) {
    __syncthreads();                       // buf[bufCur] now complete
    if (s + 1 < NSTEP) {
      load_stepB(s + 1);                   // next B globals (latency hidden)
      prefetchA(s + 1, aNext);             // next A fragments from L2
    }
    FragU bF[4];
    const uint4* bbuf = ldsB + bufCur * 512;
    #pragma unroll
    for (int j = 0; j < 4; ++j) {
      const int rec = ((wave_n * 4 + j) * 32 + lane) * 2;
      bF[j].u[0] = bbuf[rec]; bF[j].u[1] = bbuf[rec + 1];
    }
    #pragma unroll
    for (int i = 0; i < 4; ++i)
      #pragma unroll
      for (int j = 0; j < 4; ++j)
        acc[i][j] = __builtin_amdgcn_wmma_f32_16x16x32_bf16(
            false, aCur[i].h, false, bF[j].h, (short)0, acc[i][j], false, false);
    if (s + 1 < NSTEP) storeB(bufCur ^ 1); // fill other buffer for next step
  };

  load_stepB(0);
  prefetchA(0, aF0);
  storeB(0);
  for (int s = 0; s < NSTEP; s += 2) {     // NSTEP even: buffers alternate
    body(s,     aF0, aF1, 0);
    body(s + 1, aF1, aF0, 1);
  }

  // ---- epilogue: gate by (k_bucket[o] == q_bucket[n]) and store f32 -------
  int kbv[4][8];
  #pragma unroll
  for (int i = 0; i < 4; ++i) {
    const int mbase = m0 + wave_m * 64 + i * 16 + ((lane >> 4) << 3);
    #pragma unroll
    for (int v = 0; v < 8; ++v) kbv[i][v] = kb[mbase + v];
  }
  #pragma unroll
  for (int j = 0; j < 4; ++j) {
    const int pcol = p0 + wave_n * 64 + j * 16 + (lane & 15);
    const int ni   = pcol / CHW;
    const int ohw  = pcol - ni * CHW;
    const int qv   = qb[ni];
    float* obase = out + (size_t)ni * CO * CHW + ohw;
    #pragma unroll
    for (int i = 0; i < 4; ++i) {
      const int mbase = m0 + wave_m * 64 + i * 16 + ((lane >> 4) << 3);
      #pragma unroll
      for (int v = 0; v < 8; ++v) {
        const float scale = (kbv[i][v] == qv) ? 1.0f : 0.0f;
        obase[(size_t)(mbase + v) * CHW] = acc[i][j][v] * scale;
      }
    }
  }
}

// ---------------------------------------------------------------------------

extern "C" void kernel_launch(void* const* d_in, const int* in_sizes, int n_in,
                              void* d_out, int out_size, void* d_ws, size_t ws_size,
                              hipStream_t stream) {
  const float* x      = (const float*)d_in[0];   // (32,256,56,56)
  const float* kern   = (const float*)d_in[1];   // (512,2304)
  const float* hash_a = (const float*)d_in[2];   // (2307,)
  // d_in[3] = mode (unused)
  float* out = (float*)d_out;                    // (32,512,56,56)

  int*   kbuf = (int*)d_ws;                              // 512 ints
  int*   qbuf = kbuf + CO;                               // 32 ints
  uint4* wsA4 = (uint4*)((char*)d_ws + 4096);            // 2.36 MB bf16 A

  aprep_kernel<<<NMT * NSTEP * 32 / 256, 256, 0, stream>>>(kern, wsA4);
  kbucket_kernel<<<CO, 256, 0, stream>>>(kern, hash_a, kbuf);
  qbucket_kernel<<<CN, 256, 0, stream>>>(x, hash_a, qbuf);

  dim3 grid(CN * CHW / 128, CO / 256);                   // (784, 2)
  alsh_conv_wmma<<<grid, 256, 0, stream>>>(x, wsA4, kbuf, qbuf, out);
}
